// Qwen3CoderNextMoE_360777253295
// MI455X (gfx1250) — matrix operand order repaired and checked by reading, not compiled, not run on hardware
//
#include <hip/hip_runtime.h>

#define T_TOK 4096
#define DIM   1024
#define NE    512
#define TOPK  10
#define CAP   192
#define FF    256

typedef __attribute__((ext_vector_type(16))) __bf16 v16bf;
typedef __attribute__((ext_vector_type(8)))  float  v8f;
typedef __attribute__((ext_vector_type(4)))  unsigned int u32x4;
typedef __attribute__((ext_vector_type(8)))  int i32x8;
typedef __attribute__((ext_vector_type(4)))  int i32x4;

constexpr int BSTR = 33;   // f32 LDS row stride: 32 data DWORDs + 1 pad DWORD (TDM pad)
constexpr int STRH = 258;  // bf16 LDS stride for h tile
constexpr int ASTR = 36;   // bf16 LDS stride for A tile

__device__ __forceinline__ v8f wmma_bf16(v16bf a, v16bf b, v8f c) {
  // D = A(16x32 bf16) * B(32x16 bf16) + C(16x16 f32)
  return __builtin_amdgcn_wmma_f32_16x16x32_bf16(false, a, false, b, (short)0, c, false, false);
}

__device__ __forceinline__ v8f zero8() {
  v8f z;
  #pragma unroll
  for (int i = 0; i < 8; ++i) z[i] = 0.f;
  return z;
}

// Byte offset of an LDS object (addrspace(3) pointers are 32-bit LDS offsets).
__device__ __forceinline__ unsigned lds_addr(const void* p) {
  return (unsigned)(unsigned long long)(const __attribute__((address_space(3))) void*)p;
}

// Tensor Data Mover: DMA a [rows x 32] f32 tile (memory row stride = `stride` elems)
// into LDS, padding +1 DWORD per 32 DWORDs so LDS row stride = 33 DWORDs (odd ->
// bank-conflict-free fragment reads). Single instruction, tracked by TENSORcnt.
__device__ __forceinline__ void tdm_load_tile(const float* gptr, unsigned lds_byte,
                                              unsigned rows, unsigned td0, unsigned td1,
                                              unsigned stride) {
  unsigned long long ga = (unsigned long long)(size_t)gptr;
  u32x4 g0;
  g0[0] = 1u;                                                 // count=1: valid user D#
  g0[1] = lds_byte;                                           // lds_addr
  g0[2] = (unsigned)ga;                                       // global_addr[31:0]
  g0[3] = ((unsigned)(ga >> 32) & 0x1ffffffu) | (2u << 30);   // addr[56:32] | type=2
  i32x8 g1;
  g1[0] = (int)((2u << 16) | (1u << 20) | (4u << 22));        // data_size=4B, pad_en, every 32 DW, +1 DW
  g1[1] = (int)((td0 & 0xffffu) << 16);                       // tensor_dim0 lo16 (atomic barrier off)
  g1[2] = (int)(((td0 >> 16) & 0xffffu) | ((td1 & 0xffffu) << 16));
  g1[3] = (int)(((td1 >> 16) & 0xffffu) | (32u << 16));       // tile_dim0 = 32 elems
  g1[4] = (int)(rows & 0xffffu);                              // tile_dim1 = rows, tile_dim2 = 0
  g1[5] = (int)stride;                                        // tensor_dim0_stride[31:0]
  g1[6] = 0;                                                  // stride hi, dim1_stride lo
  g1[7] = 0;
  i32x4 z4; z4[0] = 0; z4[1] = 0; z4[2] = 0; z4[3] = 0;
  i32x8 z8;
  #pragma unroll
  for (int i = 0; i < 8; ++i) z8[i] = 0;
  // 6-arg form on this toolchain: (group0, group1, group2, group3, extra, cpol)
  __builtin_amdgcn_tensor_load_to_lds(g0, g1, z4, z4, z8, 0);
}

// A fragment: 16x32 tile, rows mt*16.., bf16 smem [m][k].
// Lanes 0-15 hold K 0..15, lanes 16-31 hold K 16..31 (matches frag_b_f32).
__device__ __forceinline__ v16bf frag_a(const __bf16* smem, int stride, int mt, int lane) {
  int m = mt * 16 + (lane & 15);
  int half = lane >> 4;
  const __bf16* p = smem + m * stride;
  v16bf a;
  #pragma unroll
  for (int v = 0; v < 8; ++v) {
    int kb = half * 16 + 2 * v;
    a[2 * v]     = p[kb];
    a[2 * v + 1] = p[kb + 1];
  }
  return a;
}

// B fragment from an f32 LDS tile stored [n][k] (weight-major), converting to bf16.
// Lanes 0-15 = cols with K 0..15; lanes 16-31 = cols with K 16..31.
__device__ __forceinline__ v16bf frag_b_f32(const float* smem, int stride, int nt, int lane) {
  int n = nt * 16 + (lane & 15);
  int half = lane >> 4;
  const float* p = smem + n * stride;
  v16bf b;
  #pragma unroll
  for (int v = 0; v < 8; ++v) {
    int k = half * 16 + 2 * v;
    b[2 * v]     = (__bf16)p[k];
    b[2 * v + 1] = (__bf16)p[k + 1];
  }
  return b;
}

// ------------------------------------------------- router logits (WMMA) -----
__global__ __launch_bounds__(256, 1) void moe_router_gemm(
    const float* __restrict__ x, const float* __restrict__ gw,
    float* __restrict__ logits) {
  __shared__ float  s_b[2][NE * BSTR];      // TDM-filled fp32 gate_w tiles (512 rows)
  __shared__ __bf16 s_a[32][ASTR];
  const int tb0 = blockIdx.x * 32;
  const int wave = threadIdx.x >> 5;
  const int lane = threadIdx.x & 31;

  v8f acc[2][4];
  #pragma unroll
  for (int mt = 0; mt < 2; ++mt)
    #pragma unroll
    for (int nl = 0; nl < 4; ++nl) acc[mt][nl] = zero8();

  if (wave == 0) tdm_load_tile(gw, lds_addr(&s_b[0][0]), NE, DIM, NE, DIM);
  int buf = 0;
  for (int kc = 0; kc < DIM; kc += 32) {
    {
      int i = threadIdx.x;
      int row = i >> 3, k4 = (i & 7) * 4;
      float4 v = *(const float4*)(x + (size_t)(tb0 + row) * DIM + kc + k4);
      s_a[row][k4]     = (__bf16)v.x; s_a[row][k4 + 1] = (__bf16)v.y;
      s_a[row][k4 + 2] = (__bf16)v.z; s_a[row][k4 + 3] = (__bf16)v.w;
    }
    if (wave == 0) {
      if (kc + 32 < DIM) {
        tdm_load_tile(gw + kc + 32, lds_addr(&s_b[buf ^ 1][0]), NE, DIM, NE, DIM);
        __builtin_amdgcn_s_wait_tensorcnt((short)1);
      } else {
        __builtin_amdgcn_s_wait_tensorcnt((short)0);
      }
    }
    __syncthreads();
    v16bf af[2];
    #pragma unroll
    for (int mt = 0; mt < 2; ++mt) af[mt] = frag_a(&s_a[0][0], ASTR, mt, lane);
    #pragma unroll
    for (int nl = 0; nl < 4; ++nl) {
      int nt = wave * 4 + nl;
      v16bf b = frag_b_f32(&s_b[buf][0], BSTR, nt, lane);
      acc[0][nl] = wmma_bf16(af[0], b, acc[0][nl]);
      acc[1][nl] = wmma_bf16(af[1], b, acc[1][nl]);
    }
    __syncthreads();
    buf ^= 1;
  }
  #pragma unroll
  for (int mt = 0; mt < 2; ++mt)
    #pragma unroll
    for (int nl = 0; nl < 4; ++nl)
      #pragma unroll
      for (int v = 0; v < 8; ++v) {
        int m = mt * 16 + v + 8 * (lane >> 4);
        int col = (wave * 4 + nl) * 16 + (lane & 15);
        logits[(size_t)(tb0 + m) * NE + col] = acc[mt][nl][v];
      }
}

// ------------------------------------------------- top-k + softmax ----------
__global__ void moe_topk(const float* __restrict__ logits,
                         int* __restrict__ sel, float* __restrict__ rw) {
  __shared__ float slog[NE];
  __shared__ float sval[NE];
  __shared__ int   sidx[NE];
  __shared__ float s_top[TOPK];
  __shared__ int   s_sel[TOPK];
  const int t = blockIdx.x, tid = threadIdx.x;
  slog[tid] = logits[(size_t)t * NE + tid];
  __syncthreads();
  for (int kk = 0; kk < TOPK; ++kk) {
    sval[tid] = slog[tid]; sidx[tid] = tid;
    __syncthreads();
    for (int s = NE / 2; s > 0; s >>= 1) {
      if (tid < s && sval[tid + s] > sval[tid]) { sval[tid] = sval[tid + s]; sidx[tid] = sidx[tid + s]; }
      __syncthreads();
    }
    if (tid == 0) { int b = sidx[0]; s_sel[kk] = b; s_top[kk] = sval[0]; slog[b] = -3.0e38f; }
    __syncthreads();
  }
  if (tid == 0) {
    float mx = s_top[0], sum = 0.f, ev[TOPK];
    for (int k = 0; k < TOPK; ++k) { ev[k] = __expf(s_top[k] - mx); sum += ev[k]; }
    float inv = 1.f / sum;
    for (int k = 0; k < TOPK; ++k) { rw[t * TOPK + k] = ev[k] * inv; sel[t * TOPK + k] = s_sel[k]; }
  }
}

// -------------------------------------------------------------- dispatch ----
__global__ void moe_zero_cnt(int* __restrict__ cnt) { cnt[threadIdx.x] = 0; }

__global__ void moe_build_dispatch(const int* __restrict__ sel, int* __restrict__ cnt,
                                   int* __restrict__ rowmap) {
  int r = blockIdx.x * 256 + threadIdx.x;
  if (r >= T_TOK * TOPK) return;
  int e = sel[r];
  int p = atomicAdd(&cnt[e], 1);
  if (p < CAP) rowmap[e * CAP + p] = r;
}

// ----------------------------------------------------------- expert gemm ----
__global__ __launch_bounds__(256, 1) void moe_expert_gemm(
    const float* __restrict__ x, const float* __restrict__ Wg,
    const float* __restrict__ Wu, const float* __restrict__ Wd,
    const int* __restrict__ cnt, const int* __restrict__ rowmap,
    float* __restrict__ aout) {
  __shared__ float  s_bg32[2][FF * BSTR];   // TDM-filled fp32 gate-proj tiles
  __shared__ float  s_bu32[2][FF * BSTR];   // TDM-filled fp32 up-proj tiles
  __shared__ __bf16 s_a[32][ASTR];
  __shared__ __bf16 s_h[32][STRH];
  __shared__ int s_row[32];
  __shared__ int s_tok[32];

  const int e = blockIdx.x;
  int n = cnt[e]; if (n > CAP) n = CAP;
  if (n == 0) return;
  const int wave = threadIdx.x >> 5;
  const int lane = threadIdx.x & 31;
  const float* wg = Wg + (size_t)e * FF * DIM;
  const float* wu = Wu + (size_t)e * FF * DIM;
  const float* wd = Wd + (size_t)e * DIM * FF;

  for (int rb = 0; rb < n; rb += 32) {
    if (threadIdx.x < 32) {
      int idx = rb + threadIdx.x;
      int r = (idx < n) ? rowmap[e * CAP + idx] : -1;
      s_row[threadIdx.x] = r;
      s_tok[threadIdx.x] = (r >= 0) ? (r / TOPK) : -1;
    }
    v8f accg[2][2], accu[2][2];
    #pragma unroll
    for (int mt = 0; mt < 2; ++mt)
      #pragma unroll
      for (int nl = 0; nl < 2; ++nl) { accg[mt][nl] = zero8(); accu[mt][nl] = zero8(); }
    __syncthreads();

    // ---- phase 1: g = x.Wg^T, u = x.Wu^T; TDM double-buffered weight tiles ----
    if (wave == 0) {
      tdm_load_tile(wg, lds_addr(&s_bg32[0][0]), FF, DIM, FF, DIM);
      tdm_load_tile(wu, lds_addr(&s_bu32[0][0]), FF, DIM, FF, DIM);
    }
    int buf = 0;
    for (int kc = 0; kc < DIM; kc += 32) {
      {  // stage gathered activations as bf16 (one float4 per thread)
        int i = threadIdx.x;
        int row = i >> 3, k4 = (i & 7) * 4;
        float4 v = make_float4(0.f, 0.f, 0.f, 0.f);
        int tk = s_tok[row];
        if (tk >= 0) {
          v = *(const float4*)(x + (size_t)tk * DIM + kc + k4);
          if (kc + 32 < DIM) __builtin_prefetch(x + (size_t)tk * DIM + kc + 32 + k4, 0, 1);
        }
        s_a[row][k4]     = (__bf16)v.x; s_a[row][k4 + 1] = (__bf16)v.y;
        s_a[row][k4 + 2] = (__bf16)v.z; s_a[row][k4 + 3] = (__bf16)v.w;
      }
      if (wave == 0) {
        if (kc + 32 < DIM) {  // overlap: issue next tiles, wait only for current pair
          tdm_load_tile(wg + kc + 32, lds_addr(&s_bg32[buf ^ 1][0]), FF, DIM, FF, DIM);
          tdm_load_tile(wu + kc + 32, lds_addr(&s_bu32[buf ^ 1][0]), FF, DIM, FF, DIM);
          __builtin_amdgcn_s_wait_tensorcnt((short)2);
        } else {
          __builtin_amdgcn_s_wait_tensorcnt((short)0);
        }
      }
      __syncthreads();
      v16bf af[2];
      #pragma unroll
      for (int mt = 0; mt < 2; ++mt) af[mt] = frag_a(&s_a[0][0], ASTR, mt, lane);
      #pragma unroll
      for (int nl = 0; nl < 2; ++nl) {
        int nt = wave * 2 + nl;
        v16bf bg = frag_b_f32(&s_bg32[buf][0], BSTR, nt, lane);
        v16bf bu = frag_b_f32(&s_bu32[buf][0], BSTR, nt, lane);
        #pragma unroll
        for (int mt = 0; mt < 2; ++mt) {
          accg[mt][nl] = wmma_bf16(af[mt], bg, accg[mt][nl]);
          accu[mt][nl] = wmma_bf16(af[mt], bu, accu[mt][nl]);
        }
      }
      __syncthreads();
      buf ^= 1;
    }

    // ---- fused SiLU: h = silu(g) * u -> bf16 LDS ----
    #pragma unroll
    for (int mt = 0; mt < 2; ++mt)
      #pragma unroll
      for (int nl = 0; nl < 2; ++nl) {
        int nt = wave * 2 + nl;
        #pragma unroll
        for (int v = 0; v < 8; ++v) {
          int m = mt * 16 + v + 8 * (lane >> 4);
          int ncol = nt * 16 + (lane & 15);
          float g = accg[mt][nl][v];
          float u = accu[mt][nl][v];
          s_h[m][ncol] = (__bf16)((g / (1.f + __expf(-g))) * u);
        }
      }
    __syncthreads();

    // ---- phase 2: eo = h.Wd^T; Wd double buffer aliases phase-1 regions ----
    if (wave == 0) tdm_load_tile(wd, lds_addr(&s_bg32[0][0]), 128, FF, 128, FF);
    int wbuf = 0;
    for (int nc = 0; nc < DIM; nc += 128) {
      v8f acc2[2];
      #pragma unroll
      for (int mt = 0; mt < 2; ++mt) acc2[mt] = zero8();
      for (int kf = 0; kf < FF; kf += 32) {
        if (wave == 0) {
          int nkf = kf + 32, nnc = nc;
          if (nkf == FF) { nkf = 0; nnc = nc + 128; }
          if (nnc < DIM) {
            unsigned dst = (wbuf == 0) ? lds_addr(&s_bu32[0][0]) : lds_addr(&s_bg32[0][0]);
            tdm_load_tile(wd + (size_t)nnc * FF + nkf, dst, 128, FF, 128, FF);
            __builtin_amdgcn_s_wait_tensorcnt((short)1);
          } else {
            __builtin_amdgcn_s_wait_tensorcnt((short)0);
          }
        }
        __syncthreads();
        const float* wsrc = (wbuf == 0) ? &s_bg32[0][0] : &s_bu32[0][0];
        v16bf b = frag_b_f32(wsrc, BSTR, wave, lane);
        #pragma unroll
        for (int mt = 0; mt < 2; ++mt) {
          v16bf a = frag_a(&s_h[0][0] + kf, STRH, mt, lane);
          acc2[mt] = wmma_bf16(a, b, acc2[mt]);
        }
        __syncthreads();
        wbuf ^= 1;
      }
      #pragma unroll
      for (int mt = 0; mt < 2; ++mt)
        #pragma unroll
        for (int v = 0; v < 8; ++v) {
          int m = mt * 16 + v + 8 * (lane >> 4);
          int r = s_row[m];
          if (r >= 0) {
            int col = nc + wave * 16 + (lane & 15);
            aout[(size_t)r * DIM + col] = acc2[mt][v];
          }
        }
    }
    __syncthreads();
  }
}

// ------------------------------------------------------- shared expert ------
__global__ void moe_shared_gate(const float* __restrict__ x, const float* __restrict__ sgw,
                                float* __restrict__ gate) {
  int wave = threadIdx.x >> 5, lane = threadIdx.x & 31;
  int t = blockIdx.x * 8 + wave;
  const float4* xr = (const float4*)(x + (size_t)t * DIM);
  const float4* wr = (const float4*)sgw;
  float acc = 0.f;
  for (int i = lane; i < DIM / 4; i += 32) {
    float4 a = xr[i], b = wr[i];
    acc += a.x * b.x + a.y * b.y + a.z * b.z + a.w * b.w;
  }
  for (int off = 16; off > 0; off >>= 1) acc += __shfl_down(acc, off);
  if (lane == 0) gate[t] = 1.f / (1.f + __expf(-acc));
}

__global__ __launch_bounds__(256, 1) void moe_shared_gemm(
    const float* __restrict__ x, const float* __restrict__ sWg,
    const float* __restrict__ sWu, const float* __restrict__ sWd,
    const float* __restrict__ gate, float* __restrict__ out) {
  __shared__ float  s_bg32[2][FF * BSTR];
  __shared__ float  s_bu32[2][FF * BSTR];
  __shared__ __bf16 s_a[32][ASTR];
  __shared__ __bf16 s_h[32][STRH];
  __shared__ float s_gate[32];

  const int tb0 = blockIdx.x * 32;
  const int wave = threadIdx.x >> 5;
  const int lane = threadIdx.x & 31;
  if (threadIdx.x < 32) s_gate[threadIdx.x] = gate[tb0 + threadIdx.x];

  v8f accg[2][2], accu[2][2];
  #pragma unroll
  for (int mt = 0; mt < 2; ++mt)
    #pragma unroll
    for (int nl = 0; nl < 2; ++nl) { accg[mt][nl] = zero8(); accu[mt][nl] = zero8(); }
  __syncthreads();

  if (wave == 0) {
    tdm_load_tile(sWg, lds_addr(&s_bg32[0][0]), FF, DIM, FF, DIM);
    tdm_load_tile(sWu, lds_addr(&s_bu32[0][0]), FF, DIM, FF, DIM);
  }
  int buf = 0;
  for (int kc = 0; kc < DIM; kc += 32) {
    {
      int i = threadIdx.x;
      int row = i >> 3, k4 = (i & 7) * 4;
      float4 v = *(const float4*)(x + (size_t)(tb0 + row) * DIM + kc + k4);
      s_a[row][k4]     = (__bf16)v.x; s_a[row][k4 + 1] = (__bf16)v.y;
      s_a[row][k4 + 2] = (__bf16)v.z; s_a[row][k4 + 3] = (__bf16)v.w;
    }
    if (wave == 0) {
      if (kc + 32 < DIM) {
        tdm_load_tile(sWg + kc + 32, lds_addr(&s_bg32[buf ^ 1][0]), FF, DIM, FF, DIM);
        tdm_load_tile(sWu + kc + 32, lds_addr(&s_bu32[buf ^ 1][0]), FF, DIM, FF, DIM);
        __builtin_amdgcn_s_wait_tensorcnt((short)2);
      } else {
        __builtin_amdgcn_s_wait_tensorcnt((short)0);
      }
    }
    __syncthreads();
    v16bf af[2];
    #pragma unroll
    for (int mt = 0; mt < 2; ++mt) af[mt] = frag_a(&s_a[0][0], ASTR, mt, lane);
    #pragma unroll
    for (int nl = 0; nl < 2; ++nl) {
      int nt = wave * 2 + nl;
      v16bf bg = frag_b_f32(&s_bg32[buf][0], BSTR, nt, lane);
      v16bf bu = frag_b_f32(&s_bu32[buf][0], BSTR, nt, lane);
      #pragma unroll
      for (int mt = 0; mt < 2; ++mt) {
        accg[mt][nl] = wmma_bf16(af[mt], bg, accg[mt][nl]);
        accu[mt][nl] = wmma_bf16(af[mt], bu, accu[mt][nl]);
      }
    }
    __syncthreads();
    buf ^= 1;
  }

  #pragma unroll
  for (int mt = 0; mt < 2; ++mt)
    #pragma unroll
    for (int nl = 0; nl < 2; ++nl) {
      int nt = wave * 2 + nl;
      #pragma unroll
      for (int v = 0; v < 8; ++v) {
        int m = mt * 16 + v + 8 * (lane >> 4);
        int ncol = nt * 16 + (lane & 15);
        float g = accg[mt][nl][v];
        float u = accu[mt][nl][v];
        s_h[m][ncol] = (__bf16)((g / (1.f + __expf(-g))) * u);
      }
    }
  __syncthreads();

  if (wave == 0) tdm_load_tile(sWd, lds_addr(&s_bg32[0][0]), 128, FF, 128, FF);
  int wbuf = 0;
  for (int nc = 0; nc < DIM; nc += 128) {
    v8f acc2[2];
    #pragma unroll
    for (int mt = 0; mt < 2; ++mt) acc2[mt] = zero8();
    for (int kf = 0; kf < FF; kf += 32) {
      if (wave == 0) {
        int nkf = kf + 32, nnc = nc;
        if (nkf == FF) { nkf = 0; nnc = nc + 128; }
        if (nnc < DIM) {
          unsigned dst = (wbuf == 0) ? lds_addr(&s_bu32[0][0]) : lds_addr(&s_bg32[0][0]);
          tdm_load_tile(sWd + (size_t)nnc * FF + nkf, dst, 128, FF, 128, FF);
          __builtin_amdgcn_s_wait_tensorcnt((short)1);
        } else {
          __builtin_amdgcn_s_wait_tensorcnt((short)0);
        }
      }
      __syncthreads();
      const float* wsrc = (wbuf == 0) ? &s_bg32[0][0] : &s_bu32[0][0];
      v16bf b = frag_b_f32(wsrc, BSTR, wave, lane);
      #pragma unroll
      for (int mt = 0; mt < 2; ++mt) {
        v16bf a = frag_a(&s_h[0][0] + kf, STRH, mt, lane);
        acc2[mt] = wmma_bf16(a, b, acc2[mt]);
      }
      __syncthreads();
      wbuf ^= 1;
    }
    #pragma unroll
    for (int mt = 0; mt < 2; ++mt)
      #pragma unroll
      for (int v = 0; v < 8; ++v) {
        int m = mt * 16 + v + 8 * (lane >> 4);
        int col = nc + wave * 16 + (lane & 15);
        out[(size_t)(tb0 + m) * DIM + col] = acc2[mt][v] * s_gate[m];
      }
  }
}

// ---------------------------------------------------------------- combine ---
__global__ void moe_combine(const float* __restrict__ aout, const float* __restrict__ rw,
                            float* __restrict__ out) {
  int i = blockIdx.x * 256 + threadIdx.x;     // float4 index over T*D/4
  int t = i >> 8;                             // 256 float4 per token
  int c4 = (i & 255) * 4;
  float4 acc = *(float4*)(out + (size_t)t * DIM + c4);
  #pragma unroll
  for (int k = 0; k < TOPK; ++k) {
    float w = rw[t * TOPK + k];
    float4 v = *(const float4*)(aout + (size_t)(t * TOPK + k) * DIM + c4);
    acc.x += w * v.x; acc.y += w * v.y; acc.z += w * v.z; acc.w += w * v.w;
  }
  *(float4*)(out + (size_t)t * DIM + c4) = acc;
}

// ----------------------------------------------------------------- launch ---
extern "C" void kernel_launch(void* const* d_in, const int* in_sizes, int n_in,
                              void* d_out, int out_size, void* d_ws, size_t ws_size,
                              hipStream_t stream) {
  const float* x   = (const float*)d_in[0];
  const float* gw  = (const float*)d_in[1];
  const float* Wg  = (const float*)d_in[2];
  const float* Wu  = (const float*)d_in[3];
  const float* Wd  = (const float*)d_in[4];
  const float* sWg = (const float*)d_in[5];
  const float* sWu = (const float*)d_in[6];
  const float* sWd = (const float*)d_in[7];
  const float* sgw = (const float*)d_in[8];
  float* out = (float*)d_out;

  char* ws = (char*)d_ws;
  float* aout = (float*)ws;                                   // [T*K, D] f32
  size_t off = (size_t)T_TOK * TOPK * DIM * sizeof(float);
  float* logits = (float*)(ws + off); off += (size_t)T_TOK * NE * sizeof(float);
  int*   sel    = (int*)(ws + off);   off += (size_t)T_TOK * TOPK * sizeof(int);
  float* rw     = (float*)(ws + off); off += (size_t)T_TOK * TOPK * sizeof(float);
  int*   cnt    = (int*)(ws + off);   off += (size_t)NE * sizeof(int);
  int*   rowmap = (int*)(ws + off);   off += (size_t)NE * CAP * sizeof(int);
  float* gate   = (float*)(ws + off); off += (size_t)T_TOK * sizeof(float);

  moe_router_gemm<<<T_TOK / 32, 256, 0, stream>>>(x, gw, logits);
  moe_topk<<<T_TOK, NE, 0, stream>>>(logits, sel, rw);
  moe_zero_cnt<<<1, NE, 0, stream>>>(cnt);
  moe_build_dispatch<<<(T_TOK * TOPK) / 256, 256, 0, stream>>>(sel, cnt, rowmap);
  moe_expert_gemm<<<NE, 256, 0, stream>>>(x, Wg, Wu, Wd, cnt, rowmap, aout);
  moe_shared_gate<<<T_TOK / 8, 256, 0, stream>>>(x, sgw, gate);
  moe_shared_gemm<<<T_TOK / 32, 256, 0, stream>>>(x, sWg, sWu, sWd, gate, out);
  moe_combine<<<(T_TOK * DIM / 4) / 256, 256, 0, stream>>>(aout, rw, out);
}
